// TextLSTM_byMyself_77704548319827
// MI455X (gfx1250) — compile-verified, matrix operand order checked by reading, not run on hardware
//
#include <hip/hip_runtime.h>
#include <math.h>

#define B_SZ  64
#define S_LEN 256
#define EMB   1024
#define HID   1024
#define G4    4096
#define NCLS  32000

typedef __attribute__((ext_vector_type(16))) __bf16 v16bf;
typedef __attribute__((ext_vector_type(8)))  float  v8f;

// ---------- helpers ----------

static __device__ __forceinline__ __bf16 f2bf(float f) {
  // fp32 -> bf16 round-to-nearest-even (truncate path)
  unsigned int u = __builtin_bit_cast(unsigned int, f);
  unsigned int r = (u + 0x7FFFu + ((u >> 16) & 1u)) >> 16;
  unsigned short s = (unsigned short)r;
  return __builtin_bit_cast(__bf16, s);
}

static __device__ __forceinline__ float sigmoidf_(float x) {
  return 1.0f / (1.0f + __expf(-x));
}

// Load one 16x32 bf16 operand tile in WMMA A/B layout.
// "base" points at element (row0, k0) of a row-major matrix with "stride"
// elements per row.  For the B operand the matrix is stored transposed
// (N x K row-major) so the same layout applies with row == n.
// Lane l<16: row=l, K = {0..7, 16..23}; lane l+16: row=l, K = {8..15, 24..31}.
static __device__ __forceinline__ v16bf load_tile_bf16(const __bf16* base, int stride, int lane) {
  int r  = lane & 15;
  int kh = (lane & 16) ? 8 : 0;
  const __bf16* p = base + (size_t)r * stride + kh;
  union { v16bf v; uint4 u[2]; } t;
  t.u[0] = *(const uint4*)(p);        // K = kh .. kh+7
  t.u[1] = *(const uint4*)(p + 16);   // K = kh+16 .. kh+23
  return t.v;
}

static __device__ __forceinline__ v8f wmma_bf16(v16bf a, v16bf b, v8f c) {
  return __builtin_amdgcn_wmma_f32_16x16x32_bf16(false, a, false, b, (short)0, c, false, false);
}

// ---------- setup kernels ----------

// src: K x N row-major fp32 ; dst: N x K row-major bf16 (transposed, converted)
__global__ __launch_bounds__(256) void k_transpose_cvt(const float* __restrict__ src,
                                                       __bf16* __restrict__ dst,
                                                       int K, int N) {
  long long idx = (long long)blockIdx.x * 256 + threadIdx.x;
  if (idx >= (long long)K * N) return;
  int n = (int)(idx / K);
  int k = (int)(idx % K);
  dst[idx] = f2bf(src[(size_t)k * N + n]);
}

__global__ __launch_bounds__(256) void k_init_state(float* __restrict__ c,
                                                    unsigned short* __restrict__ h0) {
  int i = blockIdx.x * 256 + threadIdx.x;
  if (i < B_SZ * HID) { c[i] = 0.0f; h0[i] = 0; }
}

// ---------- phase 1: xp = gather(C,X) @ Wi + b  (M=16384, N=4096, K=1024) ----------
// block = 256 threads = 8 waves; each wave: one 16-row M-tile x four 16-col N-tiles.
// grid = (4096/64, 16384/128)

__global__ __launch_bounds__(256) void k_xproj(const int* __restrict__ Xi,
                                               const float* __restrict__ Cemb,
                                               const __bf16* __restrict__ WiT,
                                               const float* __restrict__ b_i,
                                               const float* __restrict__ b_f,
                                               const float* __restrict__ b_g,
                                               const float* __restrict__ b_o,
                                               float* __restrict__ xp) {
  int lane = threadIdx.x & 31;
  int wave = threadIdx.x >> 5;
  int m0 = blockIdx.y * 128 + wave * 16;
  int n0 = blockIdx.x * 64;

  // fused embedding gather: one row index per lane, hoisted out of K loop
  int r  = lane & 15;
  int kh = (lane & 16) ? 8 : 0;
  int M  = m0 + r;
  int s  = M >> 6;            // M / B
  int b  = M & 63;            // M % B
  int row = Xi[b * S_LEN + s];
  const float* abase = Cemb + (size_t)row * EMB + kh;

  v8f acc[4] = {};
  for (int k0 = 0; k0 < EMB; k0 += 32) {
    const float* p = abase + k0;
    float4 f0 = *(const float4*)(p);
    float4 f1 = *(const float4*)(p + 4);
    float4 f2 = *(const float4*)(p + 16);
    float4 f3 = *(const float4*)(p + 20);
    v16bf a;
    a[0]  = f2bf(f0.x); a[1]  = f2bf(f0.y); a[2]  = f2bf(f0.z); a[3]  = f2bf(f0.w);
    a[4]  = f2bf(f1.x); a[5]  = f2bf(f1.y); a[6]  = f2bf(f1.z); a[7]  = f2bf(f1.w);
    a[8]  = f2bf(f2.x); a[9]  = f2bf(f2.y); a[10] = f2bf(f2.z); a[11] = f2bf(f2.w);
    a[12] = f2bf(f3.x); a[13] = f2bf(f3.y); a[14] = f2bf(f3.z); a[15] = f2bf(f3.w);
#pragma unroll
    for (int nt = 0; nt < 4; ++nt) {
      v16bf bt = load_tile_bf16(WiT + (size_t)(n0 + nt * 16) * EMB + k0, EMB, lane);
      acc[nt] = wmma_bf16(a, bt, acc[nt]);
    }
  }

  // epilogue: + bias, store fp32 xp (time-major (S,B,4H) == M-major here)
  const float* bp = (n0 < HID) ? b_i : (n0 < 2 * HID) ? b_f : (n0 < 3 * HID) ? b_g : b_o;
  int rowOff = (lane & 16) ? 8 : 0;
#pragma unroll
  for (int nt = 0; nt < 4; ++nt) {
    int n = n0 + nt * 16 + (lane & 15);
    float bv = bp[n & (HID - 1)];
#pragma unroll
    for (int rr = 0; rr < 8; ++rr) {
      int m = m0 + rr + rowOff;
      xp[(size_t)m * G4 + n] = acc[nt][rr] + bv;
    }
  }
}

// ---------- phase 2: one LSTM timestep (split-K, LDS reduction) ----------
// g = xp[t] + h_in @ Wh ; gates ; c,h update.  M=64, N=4096 (4 gates), K=1024.
// grid = 64 blocks, each owns 16 h-columns.  8 waves:
//   mt = wave&3 (M-tile), ks = wave>>2 (K-half: 0 -> K[0,512), 1 -> K[512,1024)).
// Each wave computes the SAME (rows, h-cols) tile for all 4 gates over its
// K-half (16 WMMA-deep instead of 32).  ks=1 waves dump partials to LDS in the
// native accumulator layout; one barrier; ks=0 waves add partials and run the
// wave-local gate/c/h epilogue.  h is ping-pong buffered across launches.

__global__ __launch_bounds__(256) void k_lstm_step(const float* __restrict__ xp_t,
                                                   const __bf16* __restrict__ h_in,
                                                   const __bf16* __restrict__ WhT,
                                                   float* __restrict__ c,
                                                   __bf16* __restrict__ h_out) {
  __shared__ float red[4 * 4 * 32 * 8];   // [mt][gate][lane][8] = 16 KB

  int lane = threadIdx.x & 31;
  int wave = threadIdx.x >> 5;
  int mt = wave & 3;
  int ks = wave >> 2;
  int m0 = mt * 16;
  int hcol0 = blockIdx.x * 16;
  int kbase = ks * (HID / 2);

  v8f acc[4] = {};
  for (int kk = 0; kk < HID / 2; kk += 32) {
    int k0 = kbase + kk;
    v16bf a = load_tile_bf16(h_in + (size_t)m0 * HID + k0, HID, lane);
#pragma unroll
    for (int q = 0; q < 4; ++q) {
      v16bf bt = load_tile_bf16(WhT + (size_t)(q * HID + hcol0) * HID + k0, HID, lane);
      acc[q] = wmma_bf16(a, bt, acc[q]);
    }
  }

  if (ks == 1) {
    // store partials in native accumulator layout: [mt][q][lane][8]
#pragma unroll
    for (int q = 0; q < 4; ++q) {
      float* dst = &red[(((mt * 4) + q) * 32 + lane) * 8];
      *(float4*)(dst)     = make_float4(acc[q][0], acc[q][1], acc[q][2], acc[q][3]);
      *(float4*)(dst + 4) = make_float4(acc[q][4], acc[q][5], acc[q][6], acc[q][7]);
    }
  }
  __syncthreads();
  if (ks != 0) return;

#pragma unroll
  for (int q = 0; q < 4; ++q) {
    const float* src = &red[(((mt * 4) + q) * 32 + lane) * 8];
    float4 p0 = *(const float4*)(src);
    float4 p1 = *(const float4*)(src + 4);
    acc[q][0] += p0.x; acc[q][1] += p0.y; acc[q][2] += p0.z; acc[q][3] += p0.w;
    acc[q][4] += p1.x; acc[q][5] += p1.y; acc[q][6] += p1.z; acc[q][7] += p1.w;
  }

  int rowOff = (lane & 16) ? 8 : 0;
  int n = hcol0 + (lane & 15);
#pragma unroll
  for (int rr = 0; rr < 8; ++rr) {
    int m = m0 + rr + rowOff;
    const float* xr = xp_t + (size_t)m * G4;
    float gi = acc[0][rr] + xr[n];
    float gf = acc[1][rr] + xr[HID + n];
    float gg = acc[2][rr] + xr[2 * HID + n];
    float go = acc[3][rr] + xr[3 * HID + n];
    float it = sigmoidf_(gi);
    float ft = sigmoidf_(gf);
    float gt = tanhf(gg);
    float ot = sigmoidf_(go);
    size_t ci = (size_t)m * HID + n;
    float cv = ft * c[ci] + it * gt;
    c[ci] = cv;
    h_out[ci] = f2bf(ot * tanhf(cv));
  }
}

// ---------- phase 3: out = h @ W_out + b_out  (M=64, N=32000, K=1024) ----------
// grid = 500 blocks x 64 cols; 8 waves: mt = wave&3, n-half = wave>>2 (2 tiles each)

__global__ __launch_bounds__(256) void k_out_gemm(const __bf16* __restrict__ h,
                                                  const __bf16* __restrict__ WoT,
                                                  const float* __restrict__ b_out,
                                                  float* __restrict__ out) {
  int lane = threadIdx.x & 31;
  int wave = threadIdx.x >> 5;
  int m0 = (wave & 3) * 16;
  int n0 = blockIdx.x * 64 + (wave >> 2) * 32;

  v8f acc[2] = {};
  for (int k0 = 0; k0 < HID; k0 += 32) {
    v16bf a = load_tile_bf16(h + (size_t)m0 * HID + k0, HID, lane);
#pragma unroll
    for (int j = 0; j < 2; ++j) {
      v16bf bt = load_tile_bf16(WoT + (size_t)(n0 + j * 16) * HID + k0, HID, lane);
      acc[j] = wmma_bf16(a, bt, acc[j]);
    }
  }
  int rowOff = (lane & 16) ? 8 : 0;
#pragma unroll
  for (int j = 0; j < 2; ++j) {
    int n = n0 + j * 16 + (lane & 15);
    float bv = b_out[n];
#pragma unroll
    for (int rr = 0; rr < 8; ++rr) {
      int m = m0 + rr + rowOff;
      out[(size_t)m * NCLS + n] = acc[j][rr] + bv;
    }
  }
}

// ---------- host ----------

extern "C" void kernel_launch(void* const* d_in, const int* in_sizes, int n_in,
                              void* d_out, int out_size, void* d_ws, size_t ws_size,
                              hipStream_t stream) {
  (void)in_sizes; (void)n_in; (void)out_size; (void)ws_size;
  const int*   X     = (const int*)d_in[0];
  const float* C     = (const float*)d_in[1];
  const float* W_ii  = (const float*)d_in[2];
  const float* W_hi  = (const float*)d_in[3];
  const float* b_i   = (const float*)d_in[4];
  const float* W_if  = (const float*)d_in[5];
  const float* W_hf  = (const float*)d_in[6];
  const float* b_f   = (const float*)d_in[7];
  const float* W_ig  = (const float*)d_in[8];
  const float* W_hg  = (const float*)d_in[9];
  const float* b_g   = (const float*)d_in[10];
  const float* W_io  = (const float*)d_in[11];
  const float* W_ho  = (const float*)d_in[12];
  const float* b_o   = (const float*)d_in[13];
  const float* W_out = (const float*)d_in[14];
  const float* b_out = (const float*)d_in[15];
  float* out = (float*)d_out;

  // workspace carve-out (256B aligned)
  char* ws = (char*)d_ws;
  size_t off = 0;
  auto walloc = [&](size_t bytes) -> void* {
    void* p = ws + off;
    off += (bytes + 255) & ~(size_t)255;
    return p;
  };
  __bf16* WiT = (__bf16*)walloc((size_t)G4 * EMB * 2);     // (4H, E)  bf16
  __bf16* WhT = (__bf16*)walloc((size_t)G4 * HID * 2);     // (4H, H)  bf16
  __bf16* WoT = (__bf16*)walloc((size_t)NCLS * HID * 2);   // (NCLS,H) bf16
  float*  xp  = (float*)walloc((size_t)B_SZ * S_LEN * G4 * 4); // (S,B,4H) fp32
  __bf16* h0  = (__bf16*)walloc((size_t)B_SZ * HID * 2);
  __bf16* h1  = (__bf16*)walloc((size_t)B_SZ * HID * 2);
  float*  cst = (float*)walloc((size_t)B_SZ * HID * 4);

  // weight transpose + bf16 convert
  const float* WiSrc[4] = {W_ii, W_if, W_ig, W_io};
  const float* WhSrc[4] = {W_hi, W_hf, W_hg, W_ho};
  unsigned nblk_w = (unsigned)((EMB * HID + 255) / 256);
  for (int q = 0; q < 4; ++q) {
    k_transpose_cvt<<<nblk_w, 256, 0, stream>>>(WiSrc[q], WiT + (size_t)q * HID * EMB, EMB, HID);
    k_transpose_cvt<<<nblk_w, 256, 0, stream>>>(WhSrc[q], WhT + (size_t)q * HID * HID, HID, HID);
  }
  unsigned nblk_o = (unsigned)(((size_t)HID * NCLS + 255) / 256);
  k_transpose_cvt<<<nblk_o, 256, 0, stream>>>(W_out, WoT, HID, NCLS);

  // zero h0, c
  k_init_state<<<(B_SZ * HID + 255) / 256, 256, 0, stream>>>(cst, (unsigned short*)h0);

  // phase 1: input projection for all timesteps
  dim3 g1(G4 / 64, (B_SZ * S_LEN) / 128);
  k_xproj<<<g1, 256, 0, stream>>>(X, C, WiT, b_i, b_f, b_g, b_o, xp);

  // phase 2: 256 sequential steps, h ping-pong (t even: h0->h1, odd: h1->h0)
  __bf16* hp[2] = {h0, h1};
  for (int t = 0; t < S_LEN; ++t) {
    const float* xpt = xp + (size_t)t * B_SZ * G4;
    k_lstm_step<<<HID / 16, 256, 0, stream>>>(xpt, hp[t & 1], WhT, cst, hp[(t + 1) & 1]);
  }
  // after t=255 (odd) final h lives in h0

  // phase 3: output projection
  k_out_gemm<<<NCLS / 64, 256, 0, stream>>>(h0, WoT, b_out, out);
}
